// GraphAttentionLayer_old_37366215475452
// MI455X (gfx1250) — matrix operand order repaired and checked by reading, not compiled, not run on hardware
//
#include <hip/hip_runtime.h>

typedef __attribute__((ext_vector_type(16))) __bf16 v16bf;
typedef __attribute__((ext_vector_type(8)))  float  v8f;

#define IN_F   512
#define OUT_F  256
#define NN     2048
#define BB     8
#define ALPHA  0.2f
#define MASK_VAL (-9000000000000000.0f)

// ---------------------------------------------------------------------------
// Kernel 0: W (512x256 f32, row-major [k][n]) -> WT (256x512 bf16, [n][k])
// ---------------------------------------------------------------------------
__global__ __launch_bounds__(256) void k_wt(const float* __restrict__ W,
                                            __bf16* __restrict__ WT) {
  int idx = blockIdx.x * 256 + threadIdx.x;      // over 512*256
  int k = idx / OUT_F;
  int n = idx % OUT_F;
  WT[(size_t)n * IN_F + k] = (__bf16)W[idx];
}

// ---------------------------------------------------------------------------
// Kernel 1: h = X @ W via v_wmma_f32_16x16x32_bf16.
// Block = 8 waves computes a 16-row tile of h; wave w owns columns [32w,32w+32).
// A-tiles (X -> bf16) are built cooperatively: each wave converts ONE 32-wide
// k-chunk and stages it in LDS in the A-operand lane layout; then all waves
// consume all 8 staged chunks (2 ds_load_b128 per A). Result stored transposed
// hT[b][n][j] (bf16) so GEMM2 gets K-contiguous B operands.
// ---------------------------------------------------------------------------
__global__ __launch_bounds__(256) void k_h(const float* __restrict__ X,
                                           const __bf16* __restrict__ WT,
                                           __bf16* __restrict__ hT) {
  __shared__ __align__(16) __bf16 abuf[8 * 512];   // 8 chunks x (16x32) bf16 = 8KB

  const int itile = blockIdx.x;      // 0..127
  const int b     = blockIdx.y;      // 0..7
  const int lane  = threadIdx.x & 31;
  const int wave  = threadIdx.x >> 5;
  const int nlo   = lane & 15;
  const int hi    = lane >> 4;       // 0 or 1 (lane half)
  const int khalf = hi * 16;         // A-operand K offset per lane half
  const int kb16  = hi * 16;         // B-operand K offset per lane half
  const int mbase = hi * 8;          // C/D row offset per lane half

  const int row = itile * 16 + nlo;                       // A: M = lane%16
  const float* xrow = X + ((size_t)b * NN + row) * IN_F;

  const int n0 = wave * 32 + nlo;
  const __bf16* w0 = WT + (size_t)n0        * IN_F;
  const __bf16* w1 = WT + (size_t)(n0 + 16) * IN_F;

  v8f acc0 = {0.f,0.f,0.f,0.f,0.f,0.f,0.f,0.f};
  v8f acc1 = {0.f,0.f,0.f,0.f,0.f,0.f,0.f,0.f};

  for (int ko = 0; ko < IN_F; ko += 256) {
    // --- produce: this wave converts k-chunk [ko+32*wave, +32) of X ---
    const int kw = ko + wave * 32;
    float tmp[16];
    *(float4*)&tmp[0]  = *(const float4*)(xrow + kw + khalf);
    *(float4*)&tmp[4]  = *(const float4*)(xrow + kw + khalf + 4);
    *(float4*)&tmp[8]  = *(const float4*)(xrow + kw + khalf + 8);
    *(float4*)&tmp[12] = *(const float4*)(xrow + kw + khalf + 12);
    v16bf A;
#pragma unroll
    for (int e = 0; e < 16; ++e) A[e] = (__bf16)tmp[e];

    __syncthreads();                         // prior reads of abuf complete
    ((v16bf*)abuf)[wave * 32 + lane] = A;
    __syncthreads();                         // all 8 chunks staged

    // --- consume: 8 chunks x 2 WMMAs, A from LDS, B from WT (L2-resident) ---
#pragma unroll
    for (int c = 0; c < 8; ++c) {
      v16bf Ac = ((v16bf*)abuf)[c * 32 + lane];
      const int kc = ko + c * 32;
      v16bf B0 = *(const v16bf*)(w0 + kc + kb16);
      acc0 = __builtin_amdgcn_wmma_f32_16x16x32_bf16(false, Ac, false, B0,
                                                     (short)0, acc0, false, false);
      v16bf B1 = *(const v16bf*)(w1 + kc + kb16);
      acc1 = __builtin_amdgcn_wmma_f32_16x16x32_bf16(false, Ac, false, B1,
                                                     (short)0, acc1, false, false);
    }
  }

  // Store transposed: hT[(b*256+n)*2048 + i]; rows m = mbase..mbase+7 contiguous
  __bf16* h0 = hT + ((size_t)b * OUT_F + n0)      * NN + itile * 16 + mbase;
  __bf16* h1 = hT + ((size_t)b * OUT_F + n0 + 16) * NN + itile * 16 + mbase;
#pragma unroll
  for (int r = 0; r < 8; ++r) h0[r] = (__bf16)acc0[r];
#pragma unroll
  for (int r = 0; r < 8; ++r) h1[r] = (__bf16)acc1[r];
}

// ---------------------------------------------------------------------------
// Kernel 2: f1[b][j] = sum_n hT[b][n][j]*a1[n]; f2 likewise with a2.
// Coalesced over j for each fixed n.
// ---------------------------------------------------------------------------
__global__ __launch_bounds__(256) void k_f(const __bf16* __restrict__ hT,
                                           const float* __restrict__ a,
                                           float* __restrict__ f1,
                                           float* __restrict__ f2) {
  int j = blockIdx.x * 256 + threadIdx.x;  // 0..2047
  int b = blockIdx.y;
  const __bf16* base = hT + (size_t)b * OUT_F * NN + j;
  float s1 = 0.f, s2 = 0.f;
#pragma unroll 4
  for (int n = 0; n < OUT_F; ++n) {
    float hv = (float)base[(size_t)n * NN];
    s1 += hv * a[n];
    s2 += hv * a[OUT_F + n];
  }
  f1[b * NN + j] = s1;
  f2[b * NN + j] = s2;
}

// ---------------------------------------------------------------------------
// Kernel 3: per-row softmax stats (max, 1/sum). One block per (b,i) row.
// Scores kept in registers (8/thread): pass 1 pure fmaxf, pass 2 one exp per
// element (halves transcendental count vs online merge).
// ---------------------------------------------------------------------------
__global__ __launch_bounds__(256) void k_stats(const int* __restrict__ adj,
                                               const float* __restrict__ f1,
                                               const float* __restrict__ f2,
                                               float* __restrict__ rowmax,
                                               float* __restrict__ rowinv) {
  const int i = blockIdx.x, b = blockIdx.y, tid = threadIdx.x;
  const int*   ar  = adj + ((size_t)b * NN + i) * NN;
  const float* f2r = f2 + b * NN;
  const float  f1v = f1[b * NN + i];

  float ev[8];
  float m = -3.0e38f;
#pragma unroll
  for (int k = 0; k < 8; ++k) {
    const int j = tid + k * 256;
    float e = f1v + f2r[j];
    e = e > 0.f ? e : ALPHA * e;
    e = ar[j] > 0 ? e : MASK_VAL;
    ev[k] = e;
    m = fmaxf(m, e);
  }

  __shared__ float sm[256];
  __shared__ float sl[256];
  sm[tid] = m;
  __syncthreads();
  for (int s = 128; s > 0; s >>= 1) {
    if (tid < s) sm[tid] = fmaxf(sm[tid], sm[tid + s]);
    __syncthreads();
  }
  m = sm[0];

  float l = 0.f;
#pragma unroll
  for (int k = 0; k < 8; ++k) l += __expf(ev[k] - m);

  sl[tid] = l;
  __syncthreads();
  for (int s = 128; s > 0; s >>= 1) {
    if (tid < s) sl[tid] = sl[tid] + sl[tid + s];
    __syncthreads();
  }
  if (tid == 0) {
    rowmax[b * NN + i] = m;
    rowinv[b * NN + i] = 1.0f / sl[0];
  }
}

// ---------------------------------------------------------------------------
// Kernel 4: h' = softmax(masked scores) @ h, WMMA bf16.
// Block = 8 waves handles a 16-row i-tile; wave w owns output cols [32w,32w+32).
// Cooperative P staging: per 256-wide j-group, wave w builds the normalized
// 16x32 P tile for chunk w (exp once per element, adj read once per block),
// stores it to LDS in A-operand lane layout; then every wave runs 8 chunks x
// 2 WMMAs with A from LDS (2x ds_load_b128) and B from hT (L2-resident).
// ---------------------------------------------------------------------------
__global__ __launch_bounds__(256) void k_attn(const int* __restrict__ adj,
                                              const float* __restrict__ f1,
                                              const float* __restrict__ f2,
                                              const float* __restrict__ rowmax,
                                              const float* __restrict__ rowinv,
                                              const __bf16* __restrict__ hT,
                                              float* __restrict__ out) {
  __shared__ __align__(16) __bf16 pbuf[8 * 512];   // 8 chunks x (16x32) bf16 = 8KB

  const int itile = blockIdx.x;     // 0..127
  const int b     = blockIdx.y;     // 0..7
  const int lane  = threadIdx.x & 31;
  const int wave  = threadIdx.x >> 5;
  const int nlo   = lane & 15;
  const int hi    = lane >> 4;
  const int khalf = hi * 16;
  const int kb16  = hi * 16;
  const int mbase = hi * 8;

  const size_t rb  = (size_t)b * NN;
  const int    row = itile * 16 + nlo;
  const float  f1v = f1[rb + row];
  const float  m   = rowmax[rb + row];
  const float  inv = rowinv[rb + row];
  const int*   ar  = adj + (rb + row) * NN;
  const float* f2r = f2 + rb;

  const int n0 = wave * 32 + nlo;
  const __bf16* h0 = hT + ((size_t)b * OUT_F + n0)      * NN;
  const __bf16* h1 = hT + ((size_t)b * OUT_F + n0 + 16) * NN;

  v8f acc0 = {0.f,0.f,0.f,0.f,0.f,0.f,0.f,0.f};
  v8f acc1 = {0.f,0.f,0.f,0.f,0.f,0.f,0.f,0.f};

  for (int jo = 0; jo < NN; jo += 256) {
    // --- produce: this wave owns j-chunk [jo+32*wave, +32) ---
    const int ja = jo + wave * 32 + khalf;   // K runs [khalf,+16)
    if (jo + 256 < NN) __builtin_prefetch(ar + ja + 256, 0, 1);

    float fv[16]; int av[16];
    *(float4*)&fv[0]  = *(const float4*)(f2r + ja);
    *(float4*)&fv[4]  = *(const float4*)(f2r + ja + 4);
    *(float4*)&fv[8]  = *(const float4*)(f2r + ja + 8);
    *(float4*)&fv[12] = *(const float4*)(f2r + ja + 12);
    *(int4*)&av[0]  = *(const int4*)(ar + ja);
    *(int4*)&av[4]  = *(const int4*)(ar + ja + 4);
    *(int4*)&av[8]  = *(const int4*)(ar + ja + 8);
    *(int4*)&av[12] = *(const int4*)(ar + ja + 12);

    v16bf A;
#pragma unroll
    for (int e = 0; e < 16; ++e) {
      float evv = f1v + fv[e];
      evv = evv > 0.f ? evv : ALPHA * evv;
      evv = av[e] > 0 ? evv : MASK_VAL;
      A[e] = (__bf16)(__expf(evv - m) * inv);   // normalized attention weight
    }

    __syncthreads();                         // prior reads of pbuf complete
    ((v16bf*)pbuf)[wave * 32 + lane] = A;
    __syncthreads();                         // all 8 chunks staged

    // --- consume: 8 chunks x 2 WMMAs ---
#pragma unroll
    for (int c = 0; c < 8; ++c) {
      v16bf Ac = ((v16bf*)pbuf)[c * 32 + lane];
      const int jc = jo + c * 32;
      v16bf B0 = *(const v16bf*)(h0 + jc + kb16);
      acc0 = __builtin_amdgcn_wmma_f32_16x16x32_bf16(false, Ac, false, B0,
                                                     (short)0, acc0, false, false);
      v16bf B1 = *(const v16bf*)(h1 + jc + kb16);
      acc1 = __builtin_amdgcn_wmma_f32_16x16x32_bf16(false, Ac, false, B1,
                                                     (short)0, acc1, false, false);
    }
  }

  float* o0 = out + ((size_t)b * NN + itile * 16 + mbase) * OUT_F + n0;
#pragma unroll
  for (int r = 0; r < 8; ++r) o0[(size_t)r * OUT_F] = acc0[r];
  float* o1 = o0 + 16;
#pragma unroll
  for (int r = 0; r < 8; ++r) o1[(size_t)r * OUT_F] = acc1[r];
}

// ---------------------------------------------------------------------------
extern "C" void kernel_launch(void* const* d_in, const int* in_sizes, int n_in,
                              void* d_out, int out_size, void* d_ws, size_t ws_size,
                              hipStream_t stream) {
  const float* X   = (const float*)d_in[0];   // (8,2048,512) f32
  const int*   adj = (const int*)  d_in[1];   // (8,2048,2048) i32
  const float* W   = (const float*)d_in[2];   // (512,256) f32
  const float* a   = (const float*)d_in[3];   // (512,1) f32
  float*       out = (float*)d_out;           // (8,2048,256) f32

  char* ws = (char*)d_ws;
  __bf16* WT = (__bf16*)ws;                                    // 256*512*2    = 262144 B
  __bf16* hT = (__bf16*)(ws + 262144);                         // 8*256*2048*2 = 8388608 B
  float* f1   = (float*)(ws + 262144 + 8388608);               // 16384 f32
  float* f2   = f1   + BB * NN;
  float* rmax = f2   + BB * NN;
  float* rinv = rmax + BB * NN;

  k_wt   <<<dim3(512),      dim3(256), 0, stream>>>(W, WT);
  k_h    <<<dim3(128, 8),   dim3(256), 0, stream>>>(X, WT, hT);
  k_f    <<<dim3(8, 8),     dim3(256), 0, stream>>>(hT, a, f1, f2);
  k_stats<<<dim3(2048, 8),  dim3(256), 0, stream>>>(adj, f1, f2, rmax, rinv);
  k_attn <<<dim3(128, 8),   dim3(256), 0, stream>>>(adj, f1, f2, rmax, rinv, hT, out);
}
